// Cluster_70050916598339
// MI455X (gfx1250) — matrix operand order repaired and checked by reading, not compiled, not run on hardware
//
#include <hip/hip_runtime.h>
#include <hip/hip_bf16.h>
#include <stddef.h>

// Problem dims (fixed by reference)
#define NB    32768      // batch
#define NC    1000       // classes
#define ND    256        // feature dim
#define NK    16         // top-K
#define NULB  50000      // ulb_dest_len

typedef __attribute__((ext_vector_type(16))) __bf16 v16bf;
typedef __attribute__((ext_vector_type(8)))  __bf16 v8bf;
typedef __attribute__((ext_vector_type(8)))  float  v8f;

__device__ __forceinline__ unsigned short f2bf(float x) {
    unsigned u = __float_as_uint(x);
    unsigned r = u + 0x7FFFu + ((u >> 16) & 1u);   // round-to-nearest-even
    return (unsigned short)(r >> 16);
}

// ---------------------------------------------------------------------------
// 0) init: zero segment stats, winner = -1, selected_label region = -1
// ---------------------------------------------------------------------------
__global__ void __launch_bounds__(256) init_kernel(float* stats /*3*NC*/,
                                                   int* winner /*NULB*/,
                                                   float* out_sel /*NULB*/) {
    int i = blockIdx.x * blockDim.x + threadIdx.x;
    if (i < 3 * NC) stats[i] = 0.0f;
    int j = i - 3 * NC;
    if (j >= 0 && j < NULB) { winner[j] = -1; out_sel[j] = -1.0f; }
}

// ---------------------------------------------------------------------------
// 1) row L2-normalize feature (rows 0..NB-1) and centroids (rows NB..NB+NC-1)
//    and convert to bf16. One block of 256 threads per 256-wide row.
// ---------------------------------------------------------------------------
__global__ void __launch_bounds__(256) norm_kernel(const float* __restrict__ feat,
                                                   const float* __restrict__ cent,
                                                   unsigned short* __restrict__ Fn,
                                                   unsigned short* __restrict__ Cn) {
    int row = blockIdx.x;
    const float* src;
    unsigned short* dst;
    if (row < NB) { src = feat + (size_t)row * ND;        dst = Fn + (size_t)row * ND; }
    else          { src = cent + (size_t)(row - NB) * ND; dst = Cn + (size_t)(row - NB) * ND; }
    int tid = threadIdx.x;
    float x = src[tid];
    __shared__ float red[256];
    red[tid] = x * x;
    __syncthreads();
    for (int s = 128; s > 0; s >>= 1) {
        if (tid < s) red[tid] += red[tid + s];
        __syncthreads();
    }
    float nrm = sqrtf(red[0]);
    float scale = 1.0f / fmaxf(nrm, 1e-8f);
    dst[tid] = f2bf(x * scale);
}

// ---------------------------------------------------------------------------
// 2) scores[i] = max softmax(pred[i,:]) = 1 / sum(exp(x - rowmax)), online pass
// ---------------------------------------------------------------------------
__global__ void __launch_bounds__(256) scores_kernel(const float* __restrict__ pred,
                                                     float* __restrict__ scores) {
    int row = blockIdx.x;
    const float* p = pred + (size_t)row * NC;
    int tid = threadIdx.x;
    float m = -3.0e38f, s = 0.0f;
    for (int c = tid; c < NC; c += 256) {
        float x = p[c];
        float M = fmaxf(m, x);
        s = s * __expf(m - M) + __expf(x - M);
        m = M;
    }
    __shared__ float sm[256], ss[256];
    sm[tid] = m; ss[tid] = s;
    __syncthreads();
    for (int st = 128; st > 0; st >>= 1) {
        if (tid < st) {
            float m2 = sm[tid + st], s2 = ss[tid + st];
            float M = fmaxf(sm[tid], m2);
            ss[tid] = ss[tid] * __expf(sm[tid] - M) + s2 * __expf(m2 - M);
            sm[tid] = M;
        }
        __syncthreads();
    }
    if (tid == 0) scores[row] = 1.0f / ss[0];
}

// ---------------------------------------------------------------------------
// 3) cos = Fn @ Cn^T via v_wmma_f32_16x16x32_bf16, fused row argmax/max,
//    atomic per-class count / sum / sum-of-squares.
//    Block = 256 thr = 8 waves; block owns 16 rows, waves split 63 N-tiles.
//    All 8 B chunks are loaded up front per tile so the 8-WMMA chain can
//    overlap the load tail (staged s_wait_loadcnt instead of drain-to-0).
// ---------------------------------------------------------------------------
__global__ void __launch_bounds__(256) cos_argmax_kernel(
        const unsigned short* __restrict__ Fnp, const unsigned short* __restrict__ Cnp,
        float* __restrict__ cosf, int* __restrict__ label,
        float* __restrict__ cnt, float* __restrict__ s1, float* __restrict__ s2) {
    const int tid   = threadIdx.x;
    const int lane  = tid & 31;
    const int wave  = tid >> 5;
    const int l16   = lane & 15;
    const int lhalf = (lane >> 4) & 1;
    const int mbase = blockIdx.x * 16;

    // ---- A fragments: row m = mbase + l16, 8 K-chunks of 32 ----
    const __bf16* Arow = (const __bf16*)Fnp + (size_t)(mbase + l16) * ND;
    v16bf afrag[8];
#pragma unroll
    for (int kc = 0; kc < 8; ++kc) {
        // K = half*16 + 0..15 (contiguous 32B)
        afrag[kc] = *(const v16bf*)(Arow + kc * 32 + lhalf * 16);
    }

    float rmax[8]; int ridx[8];
#pragma unroll
    for (int v = 0; v < 8; ++v) { rmax[v] = -3.0e38f; ridx[v] = 0x7fffffff; }

    const int NT = (NC + 15) / 16;   // 63
    for (int nt = wave; nt < NT; nt += 8) {
        int ncol = nt * 16 + l16;
        int nclamp = ncol < NC ? ncol : NC - 1;
        const __bf16* Brow = (const __bf16*)Cnp + (size_t)nclamp * ND;

        // issue all 16 b128 loads for this tile in one burst
        v16bf bfrag[8];
#pragma unroll
        for (int kc = 0; kc < 8; ++kc) {
            // B lane layout: col n = l16, K = lhalf*16 + 0..15 (contiguous 32B)
            bfrag[kc] = *(const v16bf*)(Brow + kc * 32 + lhalf * 16);
        }

        v8f acc = {};
#pragma unroll
        for (int kc = 0; kc < 8; ++kc) {
            acc = __builtin_amdgcn_wmma_f32_16x16x32_bf16(
                    false, afrag[kc], false, bfrag[kc], (short)0, acc, false, false);
        }

        if (ncol < NC) {
#pragma unroll
            for (int v = 0; v < 8; ++v) {
                float x = acc[v];
                if (x > rmax[v]) { rmax[v] = x; ridx[v] = ncol; }
            }
        }
    }

    // cross-lane argmax within each 16-lane group (first-index tie-break)
#pragma unroll
    for (int off = 8; off >= 1; off >>= 1) {
#pragma unroll
        for (int v = 0; v < 8; ++v) {
            float om = __shfl_xor(rmax[v], off, 32);
            int   oi = __shfl_xor(ridx[v], off, 32);
            if (om > rmax[v] || (om == rmax[v] && oi < ridx[v])) { rmax[v] = om; ridx[v] = oi; }
        }
    }

    __shared__ float smax[8][16];
    __shared__ int   sidx[8][16];
    if (l16 == 0) {
        // lane 0 holds rows 0..7 (lhalf=0), lane 16 holds rows 8..15 (lhalf=1)
#pragma unroll
        for (int v = 0; v < 8; ++v) {
            smax[wave][v + lhalf * 8] = rmax[v];
            sidx[wave][v + lhalf * 8] = ridx[v];
        }
    }
    __syncthreads();

    if (tid < 16) {
        float bm = smax[0][tid]; int bi = sidx[0][tid];
        for (int w = 1; w < 8; ++w) {
            float om = smax[w][tid]; int oi = sidx[w][tid];
            if (om > bm || (om == bm && oi < bi)) { bm = om; bi = oi; }
        }
        int m = mbase + tid;
        cosf[m]  = bm;
        label[m] = bi;
        atomicAdd(&cnt[bi], 1.0f);
        atomicAdd(&s1[bi], bm);
        atomicAdd(&s2[bi], bm * bm);
    }
}

// ---------------------------------------------------------------------------
// 4) per-class mean / unbiased std
// ---------------------------------------------------------------------------
__global__ void __launch_bounds__(256) stats_kernel(const float* __restrict__ cnt,
                                                    const float* __restrict__ s1,
                                                    const float* __restrict__ s2,
                                                    float* __restrict__ mean,
                                                    float* __restrict__ stdv) {
    int c = blockIdx.x * blockDim.x + threadIdx.x;
    if (c >= NC) return;
    float n  = cnt[c];
    float mu = s1[c] / fmaxf(n, 1.0f);
    float var = (s2[c] - n * mu * mu) / fmaxf(n - 1.0f, 1.0f);
    mean[c] = mu;
    stdv[c] = sqrtf(fmaxf(var, 0.0f) + 1e-12f);
}

// ---------------------------------------------------------------------------
// 5a) winner pass: deterministic last-writer-wins for duplicate indices
// ---------------------------------------------------------------------------
__global__ void __launch_bounds__(256) winner_kernel(const int* __restrict__ uidx,
                                                     int* __restrict__ winner) {
    int i = blockIdx.x * blockDim.x + threadIdx.x;
    if (i >= NB) return;
    atomicMax(&winner[uidx[i]], i);
}

// ---------------------------------------------------------------------------
// 5b) weight + label output + selected_label scatter
// ---------------------------------------------------------------------------
__global__ void __launch_bounds__(256) weight_kernel(
        const float* __restrict__ cosf, const int* __restrict__ label,
        const float* __restrict__ mean, const float* __restrict__ stdv,
        const int* __restrict__ uidx, const int* __restrict__ winner,
        float* __restrict__ out_label, float* __restrict__ out_weight,
        float* __restrict__ out_sel) {
    int i = blockIdx.x * blockDim.x + threadIdx.x;
    if (i >= NB) return;
    int   lb = label[i];
    float cf = cosf[i];
    float m  = mean[lb];
    float s  = stdv[lb];
    float z  = (cf - m) / s;
    float pdf = __expf(-0.5f * z * z) * 0.3989422804014327f / s;
    out_weight[i] = (cf < m) ? pdf : 1.0f;
    out_label[i]  = (float)lb;
    int u = uidx[i];
    if (winner[u] == i) out_sel[u] = (float)lb;
}

// ---------------------------------------------------------------------------
// 6) per-class sequential top-16 replace-min scan (matches lax.scan order)
// ---------------------------------------------------------------------------
__global__ void __launch_bounds__(256) topk_kernel(
        const float* __restrict__ predTopN, const int* __restrict__ label,
        const float* __restrict__ scores, const int* __restrict__ uidx,
        float* __restrict__ outTop, float* __restrict__ outFidx) {
    int c = blockIdx.x * blockDim.x + threadIdx.x;
    if (c >= NC) return;
    float tv[NK]; int ti[NK];
#pragma unroll
    for (int k = 0; k < NK; ++k) { tv[k] = predTopN[c * NK + k]; ti[k] = -1; }
    for (int i = 0; i < NB; ++i) {
        if (label[i] != c) continue;
        float sc = scores[i];
        int mi = 0; float mv = tv[0];
#pragma unroll
        for (int k = 1; k < NK; ++k)
            if (tv[k] < mv) { mv = tv[k]; mi = k; }   // first-min like jnp.argmin
        if (sc > mv) { tv[mi] = sc; ti[mi] = uidx[i]; }
    }
#pragma unroll
    for (int k = 0; k < NK; ++k) {
        outTop[c * NK + k]  = tv[k];
        outFidx[c * NK + k] = (float)ti[k];
    }
}

// ---------------------------------------------------------------------------
extern "C" void kernel_launch(void* const* d_in, const int* in_sizes, int n_in,
                              void* d_out, int out_size, void* d_ws, size_t ws_size,
                              hipStream_t stream) {
    const float* feature   = (const float*)d_in[0];   // [NB, ND]
    const float* pred      = (const float*)d_in[1];   // [NB, NC]
    const float* centroids = (const float*)d_in[2];   // [NC, ND]
    const float* predTopN  = (const float*)d_in[3];   // [NC, NK]
    const int*   uidx      = (const int*)d_in[4];     // [NB]

    // --- workspace layout (bytes) ---
    char* ws = (char*)d_ws;
    size_t off = 0;
    unsigned short* Fn = (unsigned short*)(ws + off); off += (size_t)NB * ND * 2; // 16 MB
    unsigned short* Cn = (unsigned short*)(ws + off); off += (size_t)NC * ND * 2;
    float* cosf   = (float*)(ws + off); off += (size_t)NB * 4;
    int*   label  = (int*)  (ws + off); off += (size_t)NB * 4;
    float* scores = (float*)(ws + off); off += (size_t)NB * 4;
    float* stats  = (float*)(ws + off); off += (size_t)3 * NC * 4; // cnt|s1|s2
    float* cnt = stats, *s1 = stats + NC, *s2 = stats + 2 * NC;
    float* mean   = (float*)(ws + off); off += (size_t)NC * 4;
    float* stdv   = (float*)(ws + off); off += (size_t)NC * 4;
    int*   winner = (int*)  (ws + off); off += (size_t)NULB * 4;

    // --- output layout (floats, concatenated in return order) ---
    float* out_label  = (float*)d_out;            // [NB]
    float* out_weight = out_label + NB;           // [NB]
    float* out_top    = out_weight + NB;          // [NC*NK]
    float* out_fidx   = out_top + NC * NK;        // [NC*NK]
    float* out_sel    = out_fidx + NC * NK;       // [NULB]

    // 0) init
    init_kernel<<<(3 * NC + NULB + 255) / 256, 256, 0, stream>>>(stats, winner, out_sel);
    // 1) normalize + bf16
    norm_kernel<<<NB + NC, 256, 0, stream>>>(feature, centroids, Fn, Cn);
    // 2) softmax-max scores
    scores_kernel<<<NB, 256, 0, stream>>>(pred, scores);
    // 3) WMMA cos GEMM + argmax + segment sums
    cos_argmax_kernel<<<NB / 16, 256, 0, stream>>>(Fn, Cn, cosf, label, cnt, s1, s2);
    // 4) per-class stats
    stats_kernel<<<(NC + 255) / 256, 256, 0, stream>>>(cnt, s1, s2, mean, stdv);
    // 5) winner pass + weight/label/selected
    winner_kernel<<<NB / 256, 256, 0, stream>>>(uidx, winner);
    weight_kernel<<<NB / 256, 256, 0, stream>>>(cosf, label, mean, stdv, uidx, winner,
                                                out_label, out_weight, out_sel);
    // 6) sequential per-class top-K
    topk_kernel<<<(NC + 255) / 256, 256, 0, stream>>>(predTopN, label, scores, uidx,
                                                      out_top, out_fidx);
    (void)in_sizes; (void)n_in; (void)out_size; (void)ws_size;
}
